// AutoRegressive_17334488006944
// MI455X (gfx1250) — matrix operand order repaired and checked by reading, not compiled, not run on hardware
//
#include <hip/hip_runtime.h>

// ---------------- problem constants ----------------
#define BB     4
#define LT_    64
#define LA_    448
#define LL     512              // LT + LA
#define CC     1024
#define HH     16
#define DH     64
#define DFF_   4096
#define NLAYERS 6
#define CODEC_ 1024
#define MM     (BB * LL)        // 2048 rows in flattened (B,L) GEMMs

typedef __bf16 bf16;
typedef __attribute__((ext_vector_type(16))) __bf16 v16bf;
typedef __attribute__((ext_vector_type(8)))  __bf16 v8bf;
typedef __attribute__((ext_vector_type(8)))  float  v8f;

// ---------------- WMMA fragment loaders ----------------
// A fragment (16x32 bf16, row-major source, leading dim lda):
// lanes 0-15 hold row M=lane, K=0..15; lanes 16-31 hold K=16..31 (matches B fragment)
static __device__ __forceinline__ v16bf load_a(const bf16* __restrict__ tile, int lda) {
  int lane = threadIdx.x & 31;
  const bf16* p = tile + (long)(lane & 15) * lda + ((lane >> 4) << 4);
  v8bf lo = *(const v8bf*)p;          // K = sel..sel+7
  v8bf hi = *(const v8bf*)(p + 8);    // K = sel+8..sel+15
  v16bf r;
#pragma unroll
  for (int i = 0; i < 8; ++i) { r[i] = lo[i]; r[i + 8] = hi[i]; }
  return r;
}

// B fragment (32x16, from weight stored as N x K row-major, leading dim ldb):
// lane n<16 holds column n, K=k0..k0+15 ; lane n+16 holds K=k0+16..k0+31 (contiguous 32B)
static __device__ __forceinline__ v16bf load_b_bf16(const bf16* __restrict__ tile, int ldb) {
  int lane = threadIdx.x & 31;
  const bf16* p = tile + (long)(lane & 15) * ldb + ((lane >> 4) << 4);
  return *(const v16bf*)p;
}

static __device__ __forceinline__ v16bf load_b_f32(const float* __restrict__ tile, int ldb) {
  int lane = threadIdx.x & 31;
  const float* p = tile + (long)(lane & 15) * ldb + ((lane >> 4) << 4);
  v16bf r;
#pragma unroll
  for (int i = 0; i < 16; ++i) r[i] = (bf16)p[i];
  return r;
}

// ---------------- batched NT GEMM: C = alpha * A(MxK) * B(NxK)^T (+bias) ----------------
// one wave (32 threads) per 32x64 C tile; K stepped by 32; 8 v_wmma per step.
// (64x64 tiling was tried and spills in the K-loop; 32x64 is the spill-free optimum.)
template <bool B_IS_F32>
__global__ void k_gemm(const bf16* __restrict__ A, long sA, int lda,
                       const void* __restrict__ Bw, long sB, int ldb,
                       float* __restrict__ Cm, long co, long ci, int cinner, int ldc,
                       const float* __restrict__ bias,
                       int K, float alpha, int relu)
{
  int lane = threadIdx.x & 31;
  int n0   = blockIdx.x * 64;
  long m0  = (long)blockIdx.y * 32;
  int z    = blockIdx.z;

  const bf16* Ab = A + (long)z * sA + m0 * lda;
  long boff = (long)z * sB + (long)n0 * ldb;
  float* Cb = Cm + (long)(z / cinner) * co + (long)(z % cinner) * ci + m0 * ldc + n0;

  v8f acc[2][4];
#pragma unroll
  for (int a_ = 0; a_ < 2; ++a_)
#pragma unroll
    for (int b_ = 0; b_ < 4; ++b_)
#pragma unroll
      for (int r = 0; r < 8; ++r) acc[a_][b_][r] = 0.f;

  for (int k0 = 0; k0 < K; k0 += 32) {
    v16bf a0 = load_a(Ab + k0, lda);
    v16bf a1 = load_a(Ab + (long)16 * lda + k0, lda);
#pragma unroll
    for (int t = 0; t < 4; ++t) {
      v16bf bf;
      if (B_IS_F32) bf = load_b_f32((const float*)Bw + boff + (long)t * 16 * ldb + k0, ldb);
      else          bf = load_b_bf16((const bf16*)Bw + boff + (long)t * 16 * ldb + k0, ldb);
      acc[0][t] = __builtin_amdgcn_wmma_f32_16x16x32_bf16(false, a0, false, bf, (short)0,
                                                          acc[0][t], false, false);
      acc[1][t] = __builtin_amdgcn_wmma_f32_16x16x32_bf16(false, a1, false, bf, (short)0,
                                                          acc[1][t], false, false);
    }
  }

  // C/D layout: VGPR r -> local row r (lanes 0-15) or r+8 (lanes 16-31); col = lane&15
  int rsel = (lane >> 4) << 3;
  int col0 = lane & 15;
#pragma unroll
  for (int mt = 0; mt < 2; ++mt)
#pragma unroll
    for (int t = 0; t < 4; ++t)
#pragma unroll
      for (int r = 0; r < 8; ++r) {
        int row = mt * 16 + r + rsel;
        int col = t * 16 + col0;
        float v = acc[mt][t][r] * alpha;
        if (bias) v += bias[n0 + col];
        if (relu) v = fmaxf(v, 0.f);
        Cb[(long)row * ldc + col] = v;
      }
}

// ---------------- pack embeddings + sinusoidal PE ----------------
__global__ void k_pack(const int* __restrict__ text, const int* __restrict__ audio,
                       const int* __restrict__ tlen, const int* __restrict__ alen,
                       const float* __restrict__ temb, const float* __restrict__ aemb,
                       float* __restrict__ x)
{
  int bl = blockIdx.x;
  int b = bl >> 9, l = bl & 511;
  int tl = tlen[b], al = alen[b];
  float* out = x + (long)bl * CC;
  const float* src = nullptr;
  float pos = 0.f;
  if (l < tl)              { src = temb + (long)text[b * LT_ + l] * CC;        pos = (float)l; }
  else if (l - tl < al)    { src = aemb + (long)audio[b * LA_ + (l - tl)] * CC; pos = (float)(l - tl); }
  int t = threadIdx.x;
#pragma unroll
  for (int p = 0; p < 4; ++p) {
    int c = t + p * 256;
    float val = 0.f;
    if (src) {
      float ang = pos * expf((float)(c & ~1) * (-9.210340371976184f / (float)CC));
      val = src[c] + ((c & 1) ? cosf(ang) : sinf(ang));
    }
    out[c] = val;
  }
}

// ---------------- fp32 -> bf16 convert ----------------
__global__ void k_cvt(const float* __restrict__ in, bf16* __restrict__ out) {
  long i = (long)blockIdx.x * 1024 + threadIdx.x;
#pragma unroll
  for (int p = 0; p < 4; ++p) out[i + p * 256] = (bf16)in[i + p * 256];
}

// ---------------- split qkv (M x 3C fp32) into per-head bf16 Q, K, V^T ----------------
__global__ void k_split(const float* __restrict__ qkv, bf16* __restrict__ Qh,
                        bf16* __restrict__ Kh, bf16* __restrict__ Vt)
{
  long idx = (long)blockIdx.x * 256 + threadIdx.x;   // over MM*CC
  int c = (int)(idx & (CC - 1));
  long bl = idx >> 10;
  int l = (int)(bl & (LL - 1));
  int b = (int)(bl >> 9);
  int h = c >> 6, d = c & 63;
  const float* q = qkv + bl * (3 * CC);
  long hb = (long)b * HH + h;
  Qh[(hb * LL + l) * DH + d] = (bf16)q[c];
  Kh[(hb * LL + l) * DH + d] = (bf16)q[CC + c];
  Vt[(hb * DH + d) * LL + l] = (bf16)q[2 * CC + c];
}

// ---------------- masked softmax over rows of (B,H,L,L) scores -> bf16 ----------------
__global__ void k_softmax(const float* __restrict__ sc, bf16* __restrict__ at,
                          const int* __restrict__ tlen, const int* __restrict__ alen,
                          int use_mask)
{
  __shared__ float red[256];
  int i = blockIdx.x & (LL - 1);
  int b = blockIdx.x >> 13;                      // / (H*L) = / 8192
  const float* row = sc + (long)blockIdx.x * LL;
  bf16* orow = at + (long)blockIdx.x * LL;
  int tl = tlen[b], il = tl + alen[b];
  int t = threadIdx.x;
  float v0 = row[t], v1 = row[t + 256];
  if (use_mask) {
    int j0 = t, j1 = t + 256;
    if ((j0 > i) && (j0 >= tl) && (i < il) && (j0 < il)) v0 = -1e30f;
    if ((j1 > i) && (j1 >= tl) && (i < il) && (j1 < il)) v1 = -1e30f;
  }
  red[t] = fmaxf(v0, v1); __syncthreads();
  for (int s = 128; s > 0; s >>= 1) { if (t < s) red[t] = fmaxf(red[t], red[t + s]); __syncthreads(); }
  float m = red[0]; __syncthreads();
  float e0 = expf(v0 - m), e1 = expf(v1 - m);
  red[t] = e0 + e1; __syncthreads();
  for (int s = 128; s > 0; s >>= 1) { if (t < s) red[t] += red[t + s]; __syncthreads(); }
  float inv = 1.f / red[0];
  orow[t] = (bf16)(e0 * inv);
  orow[t + 256] = (bf16)(e1 * inv);
}

// ---------------- y = LayerNorm(x + h) * g + b ; writes fp32 and bf16 copies ----------------
__global__ void k_add_ln(const float* __restrict__ xin, const float* __restrict__ h,
                         const float* __restrict__ g, const float* __restrict__ bt,
                         float* __restrict__ xo, bf16* __restrict__ xob)
{
  __shared__ float red[256];
  long row = blockIdx.x;
  const float* xr = xin + row * CC;
  const float* hr = h + row * CC;
  int t = threadIdx.x;
  float v[4]; float s = 0.f;
#pragma unroll
  for (int p = 0; p < 4; ++p) { v[p] = xr[t + p * 256] + hr[t + p * 256]; s += v[p]; }
  red[t] = s; __syncthreads();
  for (int q = 128; q > 0; q >>= 1) { if (t < q) red[t] += red[t + q]; __syncthreads(); }
  float mu = red[0] * (1.f / CC); __syncthreads();
  float s2 = 0.f;
#pragma unroll
  for (int p = 0; p < 4; ++p) { float d = v[p] - mu; s2 += d * d; }
  red[t] = s2; __syncthreads();
  for (int q = 128; q > 0; q >>= 1) { if (t < q) red[t] += red[t + q]; __syncthreads(); }
  float rs = rsqrtf(red[0] * (1.f / CC) + 1e-5f);
#pragma unroll
  for (int p = 0; p < 4; ++p) {
    int c = t + p * 256;
    float y = (v[p] - mu) * rs * g[c] + bt[c];
    xo[row * CC + c] = y;
    xob[row * CC + c] = (bf16)y;
  }
}

// ---------------- host-side GEMM dispatch ----------------
static void launch_gemm(hipStream_t stream, const bf16* A, long sA, int lda,
                        const void* Bw, bool b_is_f32, long sB, int ldb,
                        float* Cm, long co, long ci, int cinner, int ldc,
                        const float* bias, int M, int N, int K,
                        float alpha, int relu, int batch)
{
  dim3 g((unsigned)(N / 64), (unsigned)(M / 32), (unsigned)batch), blk(32);
  if (b_is_f32)
    k_gemm<true><<<g, blk, 0, stream>>>(A, sA, lda, Bw, sB, ldb, Cm, co, ci, cinner, ldc,
                                        bias, K, alpha, relu);
  else
    k_gemm<false><<<g, blk, 0, stream>>>(A, sA, lda, Bw, sB, ldb, Cm, co, ci, cinner, ldc,
                                         bias, K, alpha, relu);
}

extern "C" void kernel_launch(void* const* d_in, const int* in_sizes, int n_in,
                              void* d_out, int out_size, void* d_ws, size_t ws_size,
                              hipStream_t stream)
{
  (void)in_sizes; (void)n_in; (void)out_size; (void)ws_size;

  const int*   text     = (const int*)  d_in[0];
  const int*   audio    = (const int*)  d_in[1];
  const int*   tlen     = (const int*)  d_in[2];
  const int*   alen     = (const int*)  d_in[3];
  const float* temb     = (const float*)d_in[4];
  const float* aemb     = (const float*)d_in[5];
  const float* sa_qkv_w = (const float*)d_in[6];
  const float* sa_qkv_b = (const float*)d_in[7];
  const float* sa_out_w = (const float*)d_in[8];
  const float* sa_out_b = (const float*)d_in[9];
  const float* ca_qkv_w = (const float*)d_in[10];
  const float* ca_qkv_b = (const float*)d_in[11];
  const float* ca_out_w = (const float*)d_in[12];
  const float* ca_out_b = (const float*)d_in[13];
  const float* lin1_w   = (const float*)d_in[14];
  const float* lin1_b   = (const float*)d_in[15];
  const float* lin2_w   = (const float*)d_in[16];
  const float* lin2_b   = (const float*)d_in[17];
  const float* ln1_g    = (const float*)d_in[18];
  const float* ln1_b    = (const float*)d_in[19];
  const float* ln2_g    = (const float*)d_in[20];
  const float* ln2_b    = (const float*)d_in[21];
  const float* ln3_g    = (const float*)d_in[22];
  const float* ln3_b    = (const float*)d_in[23];

  char* ws = (char*)d_ws;
  const size_t MB = 1ull << 20;
  float* x      = (float*)(ws + 0);         //   8 MB  (M x C fp32)
  bf16*  xb     = (bf16*) (ws + 8  * MB);   //   4 MB  (M x C bf16)
  bf16*  memb   = (bf16*) (ws + 12 * MB);   //   4 MB  packed memory bf16
  float* qkv    = (float*)(ws + 16 * MB);   //  24 MB  (M x 3C fp32)
  bf16*  Qh     = (bf16*) (ws + 40 * MB);   //   4 MB  (B,H,L,dh)
  bf16*  Kh     = (bf16*) (ws + 44 * MB);   //   4 MB  (B,H,L,dh)
  bf16*  Vt     = (bf16*) (ws + 48 * MB);   //   4 MB  (B,H,dh,L)
  float* scores = (float*)(ws + 52 * MB);   //  64 MB  (B,H,L,L) fp32
  bf16*  attnb  = (bf16*) (ws + 116 * MB);  //  32 MB  (B,H,L,L) bf16
  float* ao     = (float*)(ws + 148 * MB);  //   8 MB  attn out fp32 (B,L,C)
  bf16*  aob    = (bf16*) (ws + 156 * MB);  //   4 MB
  float* proj   = (float*)(ws + 160 * MB);  //   8 MB
  float* hid    = scores;                   // alias (32 MB fits in 64 MB)
  bf16*  hidb   = attnb;                    // alias
  float* outp   = (float*)d_out;

  // ---- embed + pack + bf16 copies ----
  k_pack<<<dim3(MM), dim3(256), 0, stream>>>(text, audio, tlen, alen, temb, aemb, x);
  k_cvt<<<dim3((unsigned)((long)MM * CC / 1024)), dim3(256), 0, stream>>>(x, xb);
  k_cvt<<<dim3((unsigned)((long)MM * CC / 1024)), dim3(256), 0, stream>>>(x, memb);

  for (int l = 0; l < NLAYERS; ++l) {
    // ================= self attention =================
    launch_gemm(stream, xb, 0, CC, sa_qkv_w + (long)l * 3 * CC * CC, true, 0, CC,
                qkv, 0, 0, 1, 3 * CC, sa_qkv_b + (long)l * 3 * CC,
                MM, 3 * CC, CC, 1.f, 0, 1);
    k_split<<<dim3((unsigned)((long)MM * CC / 256)), dim3(256), 0, stream>>>(qkv, Qh, Kh, Vt);
    launch_gemm(stream, Qh, (long)LL * DH, DH, Kh, false, (long)LL * DH, DH,
                scores, (long)LL * LL, 0, 1, LL, nullptr,
                LL, LL, DH, 0.125f, 0, BB * HH);
    k_softmax<<<dim3(BB * HH * LL), dim3(256), 0, stream>>>(scores, attnb, tlen, alen, 1);
    launch_gemm(stream, attnb, (long)LL * LL, LL, Vt, false, (long)DH * LL, LL,
                ao, (long)LL * CC, DH, HH, CC, nullptr,
                LL, DH, LL, 1.f, 0, BB * HH);
    k_cvt<<<dim3((unsigned)((long)MM * CC / 1024)), dim3(256), 0, stream>>>(ao, aob);
    launch_gemm(stream, aob, 0, CC, sa_out_w + (long)l * CC * CC, true, 0, CC,
                proj, 0, 0, 1, CC, sa_out_b + (long)l * CC,
                MM, CC, CC, 1.f, 0, 1);
    k_add_ln<<<dim3(MM), dim3(256), 0, stream>>>(x, proj, ln1_g + (long)l * CC,
                                                 ln1_b + (long)l * CC, x, xb);

    // ================= cross attention =================
    launch_gemm(stream, xb, 0, CC, ca_qkv_w + (long)l * 3 * CC * CC, true, 0, CC,
                qkv, 0, 0, 1, 3 * CC, ca_qkv_b + (long)l * 3 * CC,
                MM, CC, CC, 1.f, 0, 1);                                   // Q from x
    launch_gemm(stream, memb, 0, CC, ca_qkv_w + (long)l * 3 * CC * CC + (long)CC * CC,
                true, 0, CC,
                qkv + CC, 0, 0, 1, 3 * CC, ca_qkv_b + (long)l * 3 * CC + CC,
                MM, 2 * CC, CC, 1.f, 0, 1);                               // K,V from memory
    k_split<<<dim3((unsigned)((long)MM * CC / 256)), dim3(256), 0, stream>>>(qkv, Qh, Kh, Vt);
    launch_gemm(stream, Qh, (long)LL * DH, DH, Kh, false, (long)LL * DH, DH,
                scores, (long)LL * LL, 0, 1, LL, nullptr,
                LL, LL, DH, 0.125f, 0, BB * HH);
    k_softmax<<<dim3(BB * HH * LL), dim3(256), 0, stream>>>(scores, attnb, tlen, alen, 0);
    launch_gemm(stream, attnb, (long)LL * LL, LL, Vt, false, (long)DH * LL, LL,
                ao, (long)LL * CC, DH, HH, CC, nullptr,
                LL, DH, LL, 1.f, 0, BB * HH);
    k_cvt<<<dim3((unsigned)((long)MM * CC / 1024)), dim3(256), 0, stream>>>(ao, aob);
    launch_gemm(stream, aob, 0, CC, ca_out_w + (long)l * CC * CC, true, 0, CC,
                proj, 0, 0, 1, CC, ca_out_b + (long)l * CC,
                MM, CC, CC, 1.f, 0, 1);
    k_add_ln<<<dim3(MM), dim3(256), 0, stream>>>(x, proj, ln2_g + (long)l * CC,
                                                 ln2_b + (long)l * CC, x, xb);

    // ================= FFN =================
    launch_gemm(stream, xb, 0, CC, lin1_w + (long)l * DFF_ * CC, true, 0, CC,
                hid, 0, 0, 1, DFF_, lin1_b + (long)l * DFF_,
                MM, DFF_, CC, 1.f, 1, 1);                                 // fused ReLU
    k_cvt<<<dim3((unsigned)((long)MM * DFF_ / 1024)), dim3(256), 0, stream>>>(hid, hidb);
    launch_gemm(stream, hidb, 0, DFF_, lin2_w + (long)l * CC * DFF_, true, 0, DFF_,
                proj, 0, 0, 1, CC, lin2_b + (long)l * CC,
                MM, CC, DFF_, 1.f, 0, 1);
    k_add_ln<<<dim3(MM), dim3(256), 0, stream>>>(x, proj, ln3_g + (long)l * CC,
                                                 ln3_b + (long)l * CC, x, xb);
  }

  // ---- final projection: x @ audio_emb^T -> (B, L, CODEC) ----
  launch_gemm(stream, xb, 0, CC, aemb, true, 0, CC,
              outp, 0, 0, 1, CODEC_, nullptr,
              MM, CODEC_, CC, 1.f, 0, 1);
}
